// TutorialLLM_75711683494012
// MI455X (gfx1250) — matrix operand order repaired and checked by reading, not compiled, not run on hardware
//
#include <hip/hip_runtime.h>

// ============================================================================
// MI455X (gfx1250) transformer forward. wave32, WMMA bf16 path.
// fp32 residual stream, bf16 activations/weights, flash-attention fused.
// GEMM staging uses the CDNA5 async global->LDS path when the toolchain
// exposes it (ASYNCcnt-tracked, VGPR-bypassing), double-buffered with one
// barrier per K-step.
// ============================================================================

#define DEV __device__ __forceinline__

typedef __attribute__((ext_vector_type(16))) __bf16 v16bf;
typedef __attribute__((ext_vector_type(8)))  float  v8f;
typedef int v4i32 __attribute__((vector_size(16)));

#if defined(__has_builtin)
#if __has_builtin(__builtin_amdgcn_global_load_async_to_lds_b128) && \
    __has_builtin(__builtin_amdgcn_s_wait_asynccnt)
#define USE_ASYNC_LDS 1
#endif
#endif
#ifndef USE_ASYNC_LDS
#define USE_ASYNC_LDS 0
#endif

union FragU { v16bf v; uint4 q[2]; };

DEV unsigned short f32_bf16(float f) {
  unsigned int u = __float_as_uint(f);
  u += 0x7FFFu + ((u >> 16) & 1u);   // round-to-nearest-even
  return (unsigned short)(u >> 16);
}

DEV v8f zero8() { v8f z = {0.f,0.f,0.f,0.f,0.f,0.f,0.f,0.f}; return z; }

// A-matrix 16x32 bf16 fragment (ISA 7.12.2): lanes 0-15 row M=lane,
// K = {kh*8..kh*8+7} and {16+kh*8..}, kh = lane>>4.
DEV v16bf load_frag_a(const unsigned short* lds, int stride, int row0, int k0) {
  const int lane = threadIdx.x & 31;
  const int m = lane & 15, kh = lane >> 4;
  const unsigned short* p = lds + (size_t)(row0 + m) * stride + k0 + kh * 8;
  FragU f;
  f.q[0] = *(const uint4*)p;          // ds_load_b128
  f.q[1] = *(const uint4*)(p + 16);
  return f.v;
}

// B-matrix 32x16 bf16 fragment, sourced from LDS holding B^T ([N][K]):
// lane n = column, K = {kh*8..kh*8+7} and {16+kh*8..}, kh = lane>>4.
DEV v16bf load_frag_b(const unsigned short* lds, int stride, int row0, int k0) {
  const int lane = threadIdx.x & 31;
  const int n = lane & 15, kh = lane >> 4;
  const unsigned short* p = lds + (size_t)(row0 + n) * stride + k0 + kh * 8;
  FragU f;
  f.q[0] = *(const uint4*)p;
  f.q[1] = *(const uint4*)(p + 16);
  return f.v;
}

#define WMMA_BF16(a, b, c) \
  __builtin_amdgcn_wmma_f32_16x16x32_bf16(false, (a), false, (b), (short)0, (c), false, false)

#if USE_ASYNC_LDS
// One 16-byte async global->LDS copy per lane (global_load_async_to_lds_b128).
// Builtin expects (v4i32 AS1*, v4i32 AS3*, imm offset, imm cpol).
DEV void async_cp16(const unsigned short* g, unsigned short* l) {
  __builtin_amdgcn_global_load_async_to_lds_b128(
      (__attribute__((address_space(1))) v4i32*)(v4i32*)(void*)const_cast<unsigned short*>(g),
      (__attribute__((address_space(3))) v4i32*)(v4i32*)(void*)l,
      0, 0);
}
#endif

// Shared WMMA tile compute: one 32-K step, wave does 32x64 of the 128x128 tile.
DEV void gemm_tile_compute(const unsigned short* at, const unsigned short* bt,
                           int wm, int wn, v8f acc[2][4]) {
  constexpr int LDT = 40;
  v16bf af[2], bfr[4];
  af[0] = load_frag_a(at, LDT, wm * 32, 0);
  af[1] = load_frag_a(at, LDT, wm * 32 + 16, 0);
  for (int n = 0; n < 4; ++n) bfr[n] = load_frag_b(bt, LDT, wn * 64 + n * 16, 0);
  for (int i = 0; i < 2; ++i)
    for (int n = 0; n < 4; ++n)
      acc[i][n] = WMMA_BF16(af[i], bfr[n], acc[i][n]);
}

// ---------------------------------------------------------------------------
// Weight convert + transpose: fp32 [K x N] (optionally head-grouped by g)
// -> bf16 [N x K].  src element (k, n) lives at (n/g)*K*g + k*g + (n%g);
// g == N reduces to plain row-major K x N.
// ---------------------------------------------------------------------------
__global__ __launch_bounds__(256)
void convert_transpose(const float* __restrict__ src, unsigned short* __restrict__ dst,
                       int K, int N, int g) {
  const int idx = blockIdx.x * 256 + threadIdx.x;
  if (idx >= K * N) return;
  const int n = idx / K, kk = idx - n * K;
  dst[idx] = f32_bf16(src[(size_t)(n / g) * K * g + (size_t)kk * g + (n % g)]);
}

// ---------------------------------------------------------------------------
// Embedding: x[b,t,:] = tok_emb[id] + pos_emb[t]   (fp32 residual stream)
// ---------------------------------------------------------------------------
__global__ __launch_bounds__(256)
void embed_kernel(const int* __restrict__ ids, const float* __restrict__ tok,
                  const float* __restrict__ pos, float* __restrict__ x) {
  constexpr int T = 1024, D = 1024;
  const int row = blockIdx.x;
  const int t = row & (T - 1);
  const int id = ids[row];
  float* xr = x + (size_t)row * D;
  const float* tr = tok + (size_t)id * D;
  const float* pr = pos + (size_t)t * D;
  for (int i = threadIdx.x; i < D; i += 256) xr[i] = tr[i] + pr[i];
}

// ---------------------------------------------------------------------------
// LayerNorm over D=1024, fp32 in -> bf16 out (feeds WMMA GEMMs)
// ---------------------------------------------------------------------------
__global__ __launch_bounds__(256)
void layernorm_bf16(const float* __restrict__ x, const float* __restrict__ g,
                    const float* __restrict__ bta, unsigned short* __restrict__ out) {
  constexpr int D = 1024;
  __shared__ float red[2][8];
  const int row = blockIdx.x, tid = threadIdx.x;
  const float* xr = x + (size_t)row * D;
  float vv[4], s = 0.f, ss = 0.f;
  for (int i = 0; i < 4; ++i) {
    vv[i] = xr[tid + i * 256];
    s += vv[i]; ss += vv[i] * vv[i];
  }
  for (int m = 16; m >= 1; m >>= 1) { s += __shfl_xor(s, m, 32); ss += __shfl_xor(ss, m, 32); }
  if ((tid & 31) == 0) { red[0][tid >> 5] = s; red[1][tid >> 5] = ss; }
  __syncthreads();
  float ts = 0.f, tss = 0.f;
  for (int w = 0; w < 8; ++w) { ts += red[0][w]; tss += red[1][w]; }
  const float mean = ts * (1.0f / D);
  const float var = tss * (1.0f / D) - mean * mean;
  const float inv = rsqrtf(var + 1e-5f);
  unsigned short* orow = out + (size_t)row * D;
  for (int i = 0; i < 4; ++i) {
    const int c = tid + i * 256;
    orow[c] = f32_bf16((vv[i] - mean) * inv * g[c] + bta[c]);
  }
}

// ---------------------------------------------------------------------------
// bf16 WMMA GEMM: C[M,N] = A[M,K] * B[K,N], B supplied transposed as Bt[N,K].
// Block 256 threads = 8 waves; tile 128x128x32; each wave: 32x64 = 8 wmmas/K-step.
// Double-buffered LDS, one barrier per K-step; async global->LDS when available.
// Epilogue: optional bias, ReLU, fp32 residual add; out bf16 or fp32.
// LDS row stride 40 (not 32) so A/B fragment ds_load_b128s hit distinct banks.
// ---------------------------------------------------------------------------
template <bool RELU, bool RESID, bool OUT32>
__global__ __launch_bounds__(256)
void gemm_bf16(const unsigned short* __restrict__ A,
               const unsigned short* __restrict__ Bt,
               const float* __restrict__ bias,
               const float* __restrict__ resid,
               float* __restrict__ Cf,
               unsigned short* __restrict__ Cb,
               int M, int N, int K) {
  constexpr int BK = 32, LDT = 40, TILE = 128 * LDT;
  __shared__ unsigned short As[2 * TILE];
  __shared__ unsigned short Bs[2 * TILE];

  const int tid = threadIdx.x;
  const int lane = tid & 31;
  const int w = tid >> 5;
  const int wm = w >> 1;   // 0..3 : 32-row band
  const int wn = w & 1;    // 0..1 : 64-col band
  const int bx = blockIdx.x, by = blockIdx.y;

  // staging: each thread copies 16 bf16 of A and of B per K-tile
  const int srow = tid >> 1;
  const int scol = (tid & 1) * 16;
  const unsigned short* ga = A  + (size_t)(by * 128 + srow) * K + scol;
  const unsigned short* gb = Bt + (size_t)(bx * 128 + srow) * K + scol;
  const int lofs = srow * LDT + scol;

  v8f acc[2][4];
  for (int i = 0; i < 2; ++i)
    for (int n = 0; n < 4; ++n) acc[i][n] = zero8();

  const int nk = K >> 5;

#if USE_ASYNC_LDS
  // Prologue: tile 0 -> buffer 0 (ASYNCcnt-tracked, bypasses VGPRs).
  async_cp16(ga,     As + lofs);
  async_cp16(ga + 8, As + lofs + 8);
  async_cp16(gb,     Bs + lofs);
  async_cp16(gb + 8, Bs + lofs + 8);
  for (int kt = 0; kt < nk; ++kt) {
    __builtin_amdgcn_s_wait_asynccnt(0);   // s_wait_asynccnt 0: tile kt landed
    __syncthreads();                       // publish; all waves past reads of buf[(kt+1)&1]
    if (kt + 1 < nk) {                     // overlap next tile with WMMAs below
      const size_t ko = (size_t)(kt + 1) * BK;
      const int nb = ((kt + 1) & 1) * TILE;
      async_cp16(ga + ko,     As + nb + lofs);
      async_cp16(ga + ko + 8, As + nb + lofs + 8);
      async_cp16(gb + ko,     Bs + nb + lofs);
      async_cp16(gb + ko + 8, Bs + nb + lofs + 8);
    }
    gemm_tile_compute(As + (kt & 1) * TILE, Bs + (kt & 1) * TILE, wm, wn, acc);
  }
#else
  // Fallback: register-staged double-buffered pipeline + global_prefetch_b8.
  uint4 ra0 = *(const uint4*)ga, ra1 = *(const uint4*)(ga + 8);
  uint4 rb0 = *(const uint4*)gb, rb1 = *(const uint4*)(gb + 8);
  for (int kt = 0; kt < nk; ++kt) {
    const int cur = (kt & 1) * TILE;
    *(uint4*)(As + cur + lofs) = ra0; *(uint4*)(As + cur + lofs + 8) = ra1;
    *(uint4*)(Bs + cur + lofs) = rb0; *(uint4*)(Bs + cur + lofs + 8) = rb1;
    __syncthreads();
    if (kt + 1 < nk) {
      const size_t ko = (size_t)(kt + 1) * BK;
      ra0 = *(const uint4*)(ga + ko); ra1 = *(const uint4*)(ga + ko + 8);
      rb0 = *(const uint4*)(gb + ko); rb1 = *(const uint4*)(gb + ko + 8);
      if (kt + 2 < nk) {
        __builtin_prefetch(ga + ko + BK, 0, 0);   // gfx1250 global_prefetch_b8
        __builtin_prefetch(gb + ko + BK, 0, 0);
      }
    }
    gemm_tile_compute(As + cur, Bs + cur, wm, wn, acc);
  }
#endif

  // Epilogue. C layout: VGPR r = row r (lanes 0-15) / r+8 (lanes 16-31), col = lane&15.
  const int hoff = (lane >> 4) * 8;
  const int cn0 = bx * 128 + wn * 64;
  const int rm0 = by * 128 + wm * 32;
  for (int n = 0; n < 4; ++n) {
    const int col = cn0 + n * 16 + (lane & 15);
    const float bv = bias ? bias[col] : 0.0f;
    for (int i = 0; i < 2; ++i)
      for (int r = 0; r < 8; ++r) {
        const int row = rm0 + i * 16 + r + hoff;
        float vv = acc[i][n][r] + bv;
        if (RELU) vv = fmaxf(vv, 0.0f);
        const size_t idx = (size_t)row * N + col;
        if (OUT32) {
          if (RESID) vv += resid[idx];
          Cf[idx] = vv;
        } else {
          Cb[idx] = f32_bf16(vv);
        }
      }
  }
}

// ---------------------------------------------------------------------------
// Fused causal flash-attention. Grid (T/64, B*H), 128 threads = 4 waves.
// Each wave owns 16 query rows; per key tile (64): S = Q K^T via WMMA,
// online softmax (row stats live in the 16-lane half owning each C row),
// O += P V via WMMA (P round-trips through LDS to reach A-fragment layout).
// q/k/v/o are bf16 [B*T, H*64].
// ---------------------------------------------------------------------------
__global__ __launch_bounds__(128)
void attn_kernel(const unsigned short* __restrict__ q,
                 const unsigned short* __restrict__ k,
                 const unsigned short* __restrict__ v,
                 unsigned short* __restrict__ o,
                 float scale) {
  constexpr int T = 1024, DM = 1024, HS = 64, LDT = 72;  // 72: bank-conflict pad
  __shared__ unsigned short Qs[64 * LDT];
  __shared__ unsigned short Ks[64 * LDT];   // K rows as-is == (K^T)^T -> B-frag source
  __shared__ unsigned short Vt[64 * LDT];   // V^T: [hs][key]
  __shared__ unsigned short Ps[64 * LDT];   // probabilities, bf16

  const int tid = threadIdx.x;
  const int lane = tid & 31;
  const int wid = tid >> 5;
  const int qt = blockIdx.x;
  const int bh = blockIdx.y;
  const int b = bh >> 4, h = bh & 15;
  const size_t headoff = (size_t)h * HS;
  const int hoff = (lane >> 4) * 8;

  {  // stage Q tile 64x64
    const int r = tid >> 1, c = (tid & 1) * 32;
    const unsigned short* src = q + (size_t)(b * T + qt * 64 + r) * DM + headoff + c;
    unsigned short* dst = Qs + r * LDT + c;
    for (int i = 0; i < 4; ++i) *(uint4*)(dst + i * 8) = *(const uint4*)(src + i * 8);
  }

  v8f oacc[4];
  for (int n = 0; n < 4; ++n) oacc[n] = zero8();
  float m_i[8], l_i[8];
  for (int r = 0; r < 8; ++r) { m_i[r] = -3.0e38f; l_i[r] = 0.0f; }

  for (int kt = 0; kt <= qt; ++kt) {
    {  // stage K tile (contiguous) and V tile transposed
      const int r = tid >> 1, c = (tid & 1) * 32;
      const unsigned short* ksrc = k + (size_t)(b * T + kt * 64 + r) * DM + headoff + c;
      unsigned short* kdst = Ks + r * LDT + c;
      for (int i = 0; i < 4; ++i) *(uint4*)(kdst + i * 8) = *(const uint4*)(ksrc + i * 8);
      const unsigned short* vsrc = v + (size_t)(b * T + kt * 64 + r) * DM + headoff + c;
      uint4 tv[4];
      for (int i = 0; i < 4; ++i) tv[i] = *(const uint4*)(vsrc + i * 8);
      const unsigned short* tvp = (const unsigned short*)tv;
      for (int e = 0; e < 32; ++e) Vt[(c + e) * LDT + r] = tvp[e];
    }
    __syncthreads();

    // S = Q K^T : 4 C-tiles (16 q-rows x 64 keys), K over hs=64 in 2 steps
    v8f s[4];
    for (int n = 0; n < 4; ++n) s[n] = zero8();
    for (int ks = 0; ks < 2; ++ks) {
      v16bf qf = load_frag_a(Qs, LDT, wid * 16, ks * 32);
      for (int n = 0; n < 4; ++n) {
        v16bf kf = load_frag_b(Ks, LDT, n * 16, ks * 32);
        s[n] = WMMA_BF16(qf, kf, s[n]);
      }
    }

    // scale + causal mask (diagonal tile only; kt<qt tiles are fully visible)
    if (kt == qt) {
      for (int n = 0; n < 4; ++n) {
        const int col = n * 16 + (lane & 15);
        for (int r = 0; r < 8; ++r) {
          float sv = s[n][r] * scale;
          if (col > wid * 16 + r + hoff) sv = -3.0e38f;
          s[n][r] = sv;
        }
      }
    } else {
      for (int n = 0; n < 4; ++n)
        for (int r = 0; r < 8; ++r) s[n][r] *= scale;
    }

    // online softmax (reduce across the 16-lane half that holds the row)
    for (int r = 0; r < 8; ++r) {
      float mx = s[0][r];
      for (int n = 1; n < 4; ++n) mx = fmaxf(mx, s[n][r]);
      for (int m = 8; m >= 1; m >>= 1) mx = fmaxf(mx, __shfl_xor(mx, m, 16));
      const float mnew = fmaxf(m_i[r], mx);
      const float corr = __expf(m_i[r] - mnew);
      float rs = 0.0f;
      for (int n = 0; n < 4; ++n) {
        const float p = __expf(s[n][r] - mnew);
        s[n][r] = p;
        rs += p;
      }
      for (int m = 8; m >= 1; m >>= 1) rs += __shfl_xor(rs, m, 16);
      l_i[r] = l_i[r] * corr + rs;
      m_i[r] = mnew;
      for (int n = 0; n < 4; ++n) oacc[n][r] *= corr;
    }

    // write P (bf16) so it can be re-read in A-fragment layout
    for (int n = 0; n < 4; ++n) {
      const int col = n * 16 + (lane & 15);
      for (int r = 0; r < 8; ++r)
        Ps[(wid * 16 + r + hoff) * LDT + col] = f32_bf16(s[n][r]);
    }
    __syncthreads();

    // O += P V   (K = 64 keys in 2 steps, N = 64 hs dims)
    for (int ks = 0; ks < 2; ++ks) {
      v16bf pf = load_frag_a(Ps, LDT, wid * 16, ks * 32);
      for (int n = 0; n < 4; ++n) {
        v16bf vf = load_frag_b(Vt, LDT, n * 16, ks * 32);
        oacc[n] = WMMA_BF16(pf, vf, oacc[n]);
      }
    }
    __syncthreads();
  }

  // finalize: O /= l, write bf16 into head-concatenated activation
  for (int r = 0; r < 8; ++r) {
    const float inv = 1.0f / l_i[r];
    const int row = qt * 64 + wid * 16 + r + hoff;
    for (int n = 0; n < 4; ++n) {
      const int col = n * 16 + (lane & 15);
      o[(size_t)(b * T + row) * DM + headoff + col] = f32_bf16(oacc[n][r] * inv);
    }
  }
}

// ============================================================================
// Host orchestration
// ============================================================================
extern "C" void kernel_launch(void* const* d_in, const int* in_sizes, int n_in,
                              void* d_out, int out_size, void* d_ws, size_t ws_size,
                              hipStream_t stream) {
  constexpr int Bn = 4, T = 1024, D = 1024, HS = 64, FF = 4096, V = 32000, L = 4;
  constexpr int MT = Bn * T;  // 4096 rows in every GEMM

  const int*   token_ids = (const int*)  d_in[0];
  const float* tok_emb   = (const float*)d_in[1];
  const float* pos_emb   = (const float*)d_in[2];
  const float* wq        = (const float*)d_in[3];
  const float* wk        = (const float*)d_in[4];
  const float* wv        = (const float*)d_in[5];
  const float* w_proj    = (const float*)d_in[6];
  const float* b_proj    = (const float*)d_in[7];
  const float* ln1_g     = (const float*)d_in[8];
  const float* ln1_b     = (const float*)d_in[9];
  const float* ln2_g     = (const float*)d_in[10];
  const float* ln2_b     = (const float*)d_in[11];
  const float* w1        = (const float*)d_in[12];
  const float* b1        = (const float*)d_in[13];
  const float* w2        = (const float*)d_in[14];
  const float* b2        = (const float*)d_in[15];
  const float* lnf_g     = (const float*)d_in[16];
  const float* lnf_b     = (const float*)d_in[17];
  const float* w_out     = (const float*)d_in[18];
  const float* b_out     = (const float*)d_in[19];
  float* logits = (float*)d_out;
  (void)in_sizes; (void)n_in; (void)out_size; (void)ws_size;

  // ---- workspace carve (~270 MB) ----
  char* wp = (char*)d_ws;
  auto take = [&](size_t bytes) -> char* {
    char* p = wp;
    wp += (bytes + 255) & ~(size_t)255;
    return p;
  };
  float*          xb  = (float*)         take((size_t)MT * D * 4);   // fp32 residual
  unsigned short* hb  = (unsigned short*)take((size_t)MT * D * 2);   // LN out (bf16)
  unsigned short* qb  = (unsigned short*)take((size_t)MT * D * 2);
  unsigned short* kb  = (unsigned short*)take((size_t)MT * D * 2);
  unsigned short* vb  = (unsigned short*)take((size_t)MT * D * 2);
  unsigned short* ob  = (unsigned short*)take((size_t)MT * D * 2);
  unsigned short* ffb = (unsigned short*)take((size_t)MT * FF * 2);
  unsigned short* wqt = (unsigned short*)take((size_t)L * D * D * 2);
  unsigned short* wkt = (unsigned short*)take((size_t)L * D * D * 2);
  unsigned short* wvt = (unsigned short*)take((size_t)L * D * D * 2);
  unsigned short* wpt = (unsigned short*)take((size_t)L * D * D * 2);
  unsigned short* w1t = (unsigned short*)take((size_t)L * FF * D * 2);
  unsigned short* w2t = (unsigned short*)take((size_t)L * D * FF * 2);
  unsigned short* wot = (unsigned short*)take((size_t)V * D * 2);

  auto cvt = [&](const float* s, unsigned short* dptr, int K, int N, int g) {
    const int total = K * N;
    convert_transpose<<<(total + 255) / 256, 256, 0, stream>>>(s, dptr, K, N, g);
  };

  // ---- weights -> bf16, transposed to [N][K] (head-grouped for wq/wk/wv) ----
  for (int l = 0; l < L; ++l) {
    cvt(wq + (size_t)l * D * D,       wqt + (size_t)l * D * D,  D,  D,  HS);
    cvt(wk + (size_t)l * D * D,       wkt + (size_t)l * D * D,  D,  D,  HS);
    cvt(wv + (size_t)l * D * D,       wvt + (size_t)l * D * D,  D,  D,  HS);
    cvt(w_proj + (size_t)l * D * D,   wpt + (size_t)l * D * D,  D,  D,  D);
    cvt(w1 + (size_t)l * D * FF,      w1t + (size_t)l * FF * D, D,  FF, FF);
    cvt(w2 + (size_t)l * FF * D,      w2t + (size_t)l * D * FF, FF, D,  D);
  }
  cvt(w_out, wot, D, V, V);

  embed_kernel<<<MT, 256, 0, stream>>>(token_ids, tok_emb, pos_emb, xb);

  // mode 0: bf16 out; 1: bf16 out + bias + relu; 2: fp32 out + bias + residual; 3: fp32 out + bias
  auto gemm = [&](const unsigned short* A, const unsigned short* Bt, const float* bias,
                  const float* resid, float* cf, unsigned short* cb,
                  int M, int N, int K, int mode) {
    dim3 grid(N / 128, M / 128);
    if (mode == 0)
      gemm_bf16<false, false, false><<<grid, 256, 0, stream>>>(A, Bt, bias, resid, cf, cb, M, N, K);
    else if (mode == 1)
      gemm_bf16<true, false, false><<<grid, 256, 0, stream>>>(A, Bt, bias, resid, cf, cb, M, N, K);
    else if (mode == 2)
      gemm_bf16<false, true, true><<<grid, 256, 0, stream>>>(A, Bt, bias, resid, cf, cb, M, N, K);
    else
      gemm_bf16<false, false, true><<<grid, 256, 0, stream>>>(A, Bt, bias, resid, cf, cb, M, N, K);
  };

  const float scale = 0.03125f;  // D**-0.5 (reference scales by dim_embed, not head_size)

  for (int l = 0; l < L; ++l) {
    layernorm_bf16<<<MT, 256, 0, stream>>>(xb, ln1_g + l * D, ln1_b + l * D, hb);
    gemm(hb, wqt + (size_t)l * D * D, nullptr, nullptr, nullptr, qb, MT, D, D, 0);
    gemm(hb, wkt + (size_t)l * D * D, nullptr, nullptr, nullptr, kb, MT, D, D, 0);
    gemm(hb, wvt + (size_t)l * D * D, nullptr, nullptr, nullptr, vb, MT, D, D, 0);
    attn_kernel<<<dim3(T / 64, Bn * 16), 128, 0, stream>>>(qb, kb, vb, ob, scale);
    gemm(ob, wpt + (size_t)l * D * D, b_proj + l * D, xb, xb, nullptr, MT, D, D, 2);
    layernorm_bf16<<<MT, 256, 0, stream>>>(xb, ln2_g + l * D, ln2_b + l * D, hb);
    gemm(hb, w1t + (size_t)l * FF * D, b1 + l * FF, nullptr, nullptr, ffb, MT, FF, D, 1);
    gemm(ffb, w2t + (size_t)l * D * FF, b2 + l * D, xb, xb, nullptr, MT, D, FF, 2);
  }

  layernorm_bf16<<<MT, 256, 0, stream>>>(xb, lnf_g, lnf_b, hb);
  gemm(hb, wot, b_out, nullptr, logits, nullptr, MT, V, D, 3);
}
